// AugmentedTransformer_87419764342877
// MI455X (gfx1250) — matrix operand order repaired and checked by reading, not compiled, not run on hardware
//
#include <hip/hip_runtime.h>
#include <hip/hip_bf16.h>

// ---------------- dims ----------------
#define BB   16
#define TT   256
#define VV   256
#define EE   256
#define HH   8
#define DD   32
#define FF   1024
#define TCC  16
#define CVV  256
#define TCP1 17

typedef unsigned short u16;
typedef __attribute__((ext_vector_type(16))) __bf16 v16bf;
typedef __attribute__((ext_vector_type(8)))  float  v8f;

__device__ inline u16 f2bf(float f) {
  unsigned u = __float_as_uint(f);
  u = (u + 0x7FFFu + ((u >> 16) & 1u)) >> 16;   // round-to-nearest-even
  return (u16)u;
}
__device__ inline float gelu_tanh(float x) {
  float x3 = x * x * x;
  return 0.5f * x * (1.f + tanhf(0.7978845608028654f * (x + 0.044715f * x3)));
}

union FragU { u16 u[16]; v16bf v; };

// ---------------- kernels ----------------

// f32 W[K][N] -> bf16 WT[N][K] (transposed, so GEMM B-operand is k-contiguous)
__global__ void k_cvt_t(const float* __restrict__ s, u16* __restrict__ d, int N, int K) {
  int i = blockIdx.x * 256 + threadIdx.x;
  if (i < N * K) {
    int n = i / K, k = i - n * K;
    d[i] = f2bf(s[(long)k * N + n]);
  }
}

// ---- FAST GEMM 16x16/wave: A[m,k] k-contiguous, B[k,n] k-contiguous.
// Full tiles only: M%16==0, N%16==0, K%32==0. Direct b128 fragment loads, no LDS.
__global__ void k_gemm_f(const u16* __restrict__ A, long sAm, long sAh, long sAb,
                         const u16* __restrict__ Bm, long sBn, long sBh, long sBb,
                         float* __restrict__ C, long sCm, long sCn, long sCh, long sCb,
                         const float* __restrict__ Res, u16* __restrict__ Cb,
                         int M, int N, int K, int nH, int act)
{
  int z = blockIdx.z;
  int hz = z % nH, bz = z / nH;
  long coff = hz * sCh + bz * sCb;
  int bm = blockIdx.y * 16, bn = blockIdx.x * 16;
  int lane = threadIdx.x, g = lane >> 4, lr = lane & 15;
  const u16* arow = A + hz * sAh + bz * sAb + (long)(bm + lr) * sAm + g * 8;
  const u16* bcol = Bm + hz * sBh + bz * sBb + (long)(bn + lr) * sBn + g * 8;
  v8f acc = {0.f, 0.f, 0.f, 0.f, 0.f, 0.f, 0.f, 0.f};

  int k0 = 0;
  for (; k0 + 64 <= K; k0 += 64) {
    FragU fa, fb, fa2, fb2;
    *(uint4*)&fa.u[0]  = *(const uint4*)(arow + k0);
    *(uint4*)&fa.u[8]  = *(const uint4*)(arow + k0 + 16);
    *(uint4*)&fb.u[0]  = *(const uint4*)(bcol + k0);
    *(uint4*)&fb.u[8]  = *(const uint4*)(bcol + k0 + 16);
    *(uint4*)&fa2.u[0] = *(const uint4*)(arow + k0 + 32);
    *(uint4*)&fa2.u[8] = *(const uint4*)(arow + k0 + 48);
    *(uint4*)&fb2.u[0] = *(const uint4*)(bcol + k0 + 32);
    *(uint4*)&fb2.u[8] = *(const uint4*)(bcol + k0 + 48);
    acc = __builtin_amdgcn_wmma_f32_16x16x32_bf16(false, fa.v,  false, fb.v,  (short)0, acc, false, false);
    acc = __builtin_amdgcn_wmma_f32_16x16x32_bf16(false, fa2.v, false, fb2.v, (short)0, acc, false, false);
  }
  if (k0 < K) {
    FragU fa, fb;
    *(uint4*)&fa.u[0] = *(const uint4*)(arow + k0);
    *(uint4*)&fa.u[8] = *(const uint4*)(arow + k0 + 16);
    *(uint4*)&fb.u[0] = *(const uint4*)(bcol + k0);
    *(uint4*)&fb.u[8] = *(const uint4*)(bcol + k0 + 16);
    acc = __builtin_amdgcn_wmma_f32_16x16x32_bf16(false, fa.v, false, fb.v, (short)0, acc, false, false);
  }

  int n = bn + lr;
#pragma unroll
  for (int r = 0; r < 8; r++) {
    int m = bm + r + g * 8;
    float val = acc[r];
    if (act == 1) val = gelu_tanh(val);
    long off = coff + (long)m * sCm + (long)n * sCn;
    if (Res) val += Res[off];
    if (C)  C[off]  = val;
    if (Cb) Cb[off] = f2bf(val);
  }
}

// ---- FAST GEMM 32x32/wave: 2x2 WMMA tiles, A/B fragments reused -> 2 loads per WMMA.
// Requires M%32==0, N%32==0, K%32==0.
__global__ void k_gemm_f2(const u16* __restrict__ A, long sAm, long sAh, long sAb,
                          const u16* __restrict__ Bm, long sBn, long sBh, long sBb,
                          float* __restrict__ C, long sCm, long sCn, long sCh, long sCb,
                          const float* __restrict__ Res, u16* __restrict__ Cb,
                          int M, int N, int K, int nH, int act)
{
  int z = blockIdx.z;
  int hz = z % nH, bz = z / nH;
  long coff = hz * sCh + bz * sCb;
  int bm = blockIdx.y * 32, bn = blockIdx.x * 32;
  int lane = threadIdx.x, g = lane >> 4, lr = lane & 15;
  const u16* arow0 = A + hz * sAh + bz * sAb + (long)(bm + lr) * sAm + g * 8;
  const u16* arow1 = arow0 + 16 * sAm;
  const u16* bcol0 = Bm + hz * sBh + bz * sBb + (long)(bn + lr) * sBn + g * 8;
  const u16* bcol1 = bcol0 + 16 * sBn;
  v8f a00 = {0.f,0.f,0.f,0.f,0.f,0.f,0.f,0.f};
  v8f a01 = a00, a10 = a00, a11 = a00;

  for (int k0 = 0; k0 < K; k0 += 32) {
    __builtin_prefetch(arow0 + k0 + 256, 0, 1);   // global_prefetch_b8, 256 elements (8 k-steps) ahead
    __builtin_prefetch(bcol0 + k0 + 256, 0, 1);
    FragU fa0, fa1, fb0, fb1;
    *(uint4*)&fa0.u[0] = *(const uint4*)(arow0 + k0);
    *(uint4*)&fa0.u[8] = *(const uint4*)(arow0 + k0 + 16);
    *(uint4*)&fa1.u[0] = *(const uint4*)(arow1 + k0);
    *(uint4*)&fa1.u[8] = *(const uint4*)(arow1 + k0 + 16);
    *(uint4*)&fb0.u[0] = *(const uint4*)(bcol0 + k0);
    *(uint4*)&fb0.u[8] = *(const uint4*)(bcol0 + k0 + 16);
    *(uint4*)&fb1.u[0] = *(const uint4*)(bcol1 + k0);
    *(uint4*)&fb1.u[8] = *(const uint4*)(bcol1 + k0 + 16);
    a00 = __builtin_amdgcn_wmma_f32_16x16x32_bf16(false, fa0.v, false, fb0.v, (short)0, a00, false, false);
    a01 = __builtin_amdgcn_wmma_f32_16x16x32_bf16(false, fa0.v, false, fb1.v, (short)0, a01, false, false);
    a10 = __builtin_amdgcn_wmma_f32_16x16x32_bf16(false, fa1.v, false, fb0.v, (short)0, a10, false, false);
    a11 = __builtin_amdgcn_wmma_f32_16x16x32_bf16(false, fa1.v, false, fb1.v, (short)0, a11, false, false);
  }

#pragma unroll
  for (int mi = 0; mi < 2; mi++) {
#pragma unroll
    for (int ni = 0; ni < 2; ni++) {
      v8f acc = (mi == 0) ? (ni == 0 ? a00 : a01) : (ni == 0 ? a10 : a11);
      int n = bn + ni * 16 + lr;
#pragma unroll
      for (int r = 0; r < 8; r++) {
        int m = bm + mi * 16 + r + g * 8;
        float val = acc[r];
        if (act == 1) val = gelu_tanh(val);
        long off = coff + (long)m * sCm + (long)n * sCn;
        if (Res) val += Res[off];
        if (C)  C[off]  = val;
        if (Cb) Cb[off] = f2bf(val);
      }
    }
  }
}

// ---- GENERIC GEMM: fully strided, edge tiles allowed (clamp + select, branch-free)
__global__ void k_gemm(const u16* __restrict__ A, long sAm, long sAk, long sAh, long sAb,
                       const u16* __restrict__ Bm, long sBk, long sBn, long sBh, long sBb,
                       float* __restrict__ C, long sCm, long sCn, long sCh, long sCb,
                       const float* __restrict__ Res, u16* __restrict__ Cb,
                       int M, int N, int K, int nH, int act)
{
  __shared__ u16 la[512];   // 16 x 32 A tile
  __shared__ u16 lb[512];   // 32 x 16 B tile
  int z  = blockIdx.z;
  int hz = z % nH, bz = z / nH;
  const u16* Ab = A + hz * sAh + bz * sAb;
  const u16* Bb = Bm + hz * sBh + bz * sBb;
  long coff = hz * sCh + bz * sCb;
  int bm = blockIdx.y * 16, bn = blockIdx.x * 16;
  int lane = threadIdx.x;
  v8f acc = {0.f, 0.f, 0.f, 0.f, 0.f, 0.f, 0.f, 0.f};

  for (int k0 = 0; k0 < K; k0 += 32) {
    for (int j = 0; j < 16; j++) {
      int flat = lane * 16 + j;
      int r = flat >> 5, c = flat & 31;             // A tile [16][32]
      int gm = bm + r, gk = k0 + c;
      int gmc = gm < M - 1 ? gm : M - 1;
      int gkc = gk < K - 1 ? gk : K - 1;
      u16 va = Ab[(long)gmc * sAm + (long)gkc * sAk];
      la[flat] = (gm < M && gk < K) ? va : (u16)0;
      int rk = flat >> 4, cn = flat & 15;           // B tile [32][16]
      int gk2 = k0 + rk, gn = bn + cn;
      int gk2c = gk2 < K - 1 ? gk2 : K - 1;
      int gnc  = gn < N - 1 ? gn : N - 1;
      u16 vbv = Bb[(long)gk2c * sBk + (long)gnc * sBn];
      lb[flat] = (gk2 < K && gn < N) ? vbv : (u16)0;
    }
    __syncthreads();
    FragU fa, fb;
    int g = lane >> 4, lr = lane & 15;
#pragma unroll
    for (int e = 0; e < 16; e++) {
      int ka = (e < 8 ? e : e + 8) + g * 8;
      fa.u[e] = la[lr * 32 + ka];
      int kb = (e < 8 ? e : e + 8) + g * 8;
      fb.u[e] = lb[kb * 16 + lr];
    }
    acc = __builtin_amdgcn_wmma_f32_16x16x32_bf16(false, fa.v, false, fb.v,
                                                  (short)0, acc, false, false);
    __syncthreads();
  }

  int g2 = lane >> 4;
  int n = bn + (lane & 15);
#pragma unroll
  for (int r = 0; r < 8; r++) {
    int m = bm + r + g2 * 8;
    if (m < M && n < N) {
      float val = acc[r];
      if (act == 1) val = gelu_tanh(val);
      long off = coff + (long)m * sCm + (long)n * sCn;
      if (Res) val += Res[off];
      if (C)  C[off]  = val;
      if (Cb) Cb[off] = f2bf(val);
    }
  }
}

// LayerNorm over E=256 per row; writes f32 and/or bf16
__global__ void k_ln(const float* __restrict__ x, const float* __restrict__ gmm,
                     float* __restrict__ oF, u16* __restrict__ oB)
{
  __shared__ float red[256];
  int row = blockIdx.x, t = threadIdx.x;
  float v = x[(long)row * EE + t];
  red[t] = v; __syncthreads();
  for (int s = 128; s > 0; s >>= 1) { if (t < s) red[t] += red[t + s]; __syncthreads(); }
  float m = red[0] * (1.f / EE); __syncthreads();
  float d = v - m;
  red[t] = d * d; __syncthreads();
  for (int s = 128; s > 0; s >>= 1) { if (t < s) red[t] += red[t + s]; __syncthreads(); }
  float var = red[0] * (1.f / EE);
  float y = d * rsqrtf(var + 1e-6f) * gmm[t];
  if (oF) oF[(long)row * EE + t] = y;
  if (oB) oB[(long)row * EE + t] = f2bf(y);
}

// softmax over rows of [nrows, L]; q = row % Tq for causal mask; writes bf16 probs
__global__ void k_softmax(const float* __restrict__ S, u16* __restrict__ P,
                          int L, int Tq, float scale, int causal)
{
  __shared__ float red[256];
  int row = blockIdx.x, t = threadIdx.x;
  int q = row % Tq;
  bool valid = (t < L) && (!causal || t <= q);
  float v = valid ? S[(long)row * L + t] * scale : -1e30f;
  red[t] = v; __syncthreads();
  for (int s = 128; s > 0; s >>= 1) { if (t < s) red[t] = fmaxf(red[t], red[t + s]); __syncthreads(); }
  float mx = red[0]; __syncthreads();
  float e = valid ? __expf(v - mx) : 0.f;
  red[t] = e; __syncthreads();
  for (int s = 128; s > 0; s >>= 1) { if (t < s) red[t] += red[t + s]; __syncthreads(); }
  float sum = red[0];
  if (t < L) P[(long)row * L + t] = f2bf(e / sum);
}

__global__ void k_embed(const int* __restrict__ ids, int idsStride, int idsOff,
                        const float* __restrict__ tokE, const float* __restrict__ posE,
                        int Tlen, float* __restrict__ oF, u16* __restrict__ oB)
{
  int row = blockIdx.x, t = threadIdx.x;
  int pos = row % Tlen, b = row / Tlen;
  int tok = ids[b * idsStride + idsOff + pos];
  float v = tokE[(long)tok * EE + t] + posE[(long)pos * EE + t];
  if (oF) oF[(long)row * EE + t] = v;
  if (oB) oB[(long)row * EE + t] = f2bf(v);
}

__global__ void k_init_ids(int* ids) {
  int t = blockIdx.x * 256 + threadIdx.x;
  if (t < BB * TCP1) ids[t] = (t % TCP1 == 0) ? CVV : 0;   // BOS id == CV
}

__global__ void k_mean(const float* __restrict__ x, u16* __restrict__ o, int Trows) {
  int b = blockIdx.x, e = threadIdx.x;
  float s = 0.f;
  for (int t = 0; t < Trows; t++) s += x[((long)b * Trows + t) * EE + e];
  o[b * EE + e] = f2bf(s / (float)Trows);
}

__global__ void k_argmax(const float* __restrict__ logits, long bStride,
                         int* __restrict__ ids, int idsStride, int idsOff,
                         float* __restrict__ outTok, int outStride)
{
  __shared__ float rv[256];
  __shared__ int   ri[256];
  int b = blockIdx.x, t = threadIdx.x;
  rv[t] = logits[(long)b * bStride + t]; ri[t] = t; __syncthreads();
  for (int s = 128; s > 0; s >>= 1) {
    if (t < s && rv[t + s] > rv[t]) { rv[t] = rv[t + s]; ri[t] = ri[t + s]; }
    __syncthreads();
  }
  if (t == 0) {
    ids[b * idsStride + idsOff] = ri[0];
    outTok[(long)b * outStride] = (float)ri[0];
  }
}

// ---------------- host orchestration ----------------

extern "C" void kernel_launch(void* const* d_in, const int* in_sizes, int n_in,
                              void* d_out, int out_size, void* d_ws, size_t ws_size,
                              hipStream_t stream)
{
  (void)in_sizes; (void)n_in; (void)out_size; (void)ws_size;

  int cur = 0;
  const int* inputs = (const int*)d_in[cur++];
  auto nf = [&]() { return (const float*)d_in[cur++]; };

  struct AttnP { const float *Wq, *Wk, *Wv, *Wo; };
  struct TLayerP { const float* ln1; AttnP a; const float* ln2; const float *W1, *W2; };
  struct CLayerP { const float* ln1; AttnP sa; const float* ln2; AttnP ca; const float* ln3; const float *W1, *W2; };
  auto rdA = [&]() { AttnP a; a.Wq = nf(); a.Wk = nf(); a.Wv = nf(); a.Wo = nf(); return a; };
  auto rdT = [&]() { TLayerP t; t.ln1 = nf(); t.a = rdA(); t.ln2 = nf(); t.W1 = nf(); t.W2 = nf(); return t; };
  auto rdC = [&]() { CLayerP c; c.ln1 = nf(); c.sa = rdA(); c.ln2 = nf(); c.ca = rdA(); c.ln3 = nf(); c.W1 = nf(); c.W2 = nf(); return c; };

  // ---- param walk (setup_inputs dict order) ----
  const float* cot_tok = nf();  const float* cot_pos = nf();
  TLayerP cin[2] = { rdT(), rdT() };
  const float* cot_ctok = nf(); const float* cot_cpos = nf();
  CLayerP ccr[2] = { rdC(), rdC() };
  const float* cot_lnf = nf();  const float* cot_head = nf();
  const float* enc_ctok = nf(); const float* enc_cpos = nf();
  TLayerP ecot[2] = { rdT(), rdT() };
  const float* enc_tok = nf();  const float* enc_pos = nf();
  CLayerP ecr[2] = { rdC(), rdC() };
  const float* enc_lnf = nf();  const float* enc_head = nf();

  // ---- workspace allocator ----
  char* wsp = (char*)d_ws;
  auto alloc = [&](size_t bytes) { void* p = wsp; wsp += (bytes + 255) & ~(size_t)255; return p; };
  auto cvtT_ = [&](const float* s, int N, int K) {
    u16* d = (u16*)alloc((size_t)N * K * 2);
    k_cvt_t<<<dim3((N * K + 255) / 256), dim3(256), 0, stream>>>(s, d, N, K);
    return d;
  };

  struct AttnB { u16 *q, *k, *v, *o; };
  auto cvtA = [&](const AttnP& a) {
    AttnB b;
    b.q = cvtT_(a.Wq, EE, EE); b.k = cvtT_(a.Wk, EE, EE);
    b.v = cvtT_(a.Wv, EE, EE); b.o = cvtT_(a.Wo, EE, EE);
    return b;
  };
  struct TLayerB { AttnB a; u16 *w1, *w2; };
  auto cvtTL = [&](const TLayerP& t) { TLayerB b; b.a = cvtA(t.a); b.w1 = cvtT_(t.W1, FF, EE); b.w2 = cvtT_(t.W2, EE, FF); return b; };
  struct CLayerB { AttnB sa, ca; u16 *w1, *w2; };
  auto cvtCL = [&](const CLayerP& c) { CLayerB b; b.sa = cvtA(c.sa); b.ca = cvtA(c.ca); b.w1 = cvtT_(c.W1, FF, EE); b.w2 = cvtT_(c.W2, EE, FF); return b; };

  TLayerB cinB[2] = { cvtTL(cin[0]), cvtTL(cin[1]) };
  CLayerB ccrB[2] = { cvtCL(ccr[0]), cvtCL(ccr[1]) };
  u16* cotHeadT = cvtT_(cot_head, CVV, EE);
  TLayerB ecotB[2] = { cvtTL(ecot[0]), cvtTL(ecot[1]) };
  CLayerB ecrB[2] = { cvtCL(ecr[0]), cvtCL(ecr[1]) };
  u16* encHeadT = cvtT_(enc_head, VV, EE);

  // ---- activation buffers ----
  const long RMAX = (long)BB * TT;                       // 4096 rows max
  float* xE   = (float*)alloc(RMAX * EE * 4);            // residual stream
  u16*  xEb   = (u16*)alloc(RMAX * EE * 2);              // bf16 of encoder x (cross K/V source)
  u16*  hb    = (u16*)alloc(RMAX * EE * 2);              // LN output (bf16)
  u16*  qb    = (u16*)alloc(RMAX * EE * 2);
  u16*  kbuf  = (u16*)alloc(RMAX * EE * 2);              // K: [B,Tk,H,D]
  u16*  vbuf  = (u16*)alloc(RMAX * EE * 2);              // V transposed: [B,H*D,Tk]
  u16*  obuf  = (u16*)alloc(RMAX * EE * 2);              // O: [B,Tq,H,D]
  u16*  hid   = (u16*)alloc(RMAX * FF * 2);              // MLP hidden (bf16, GELU applied)
  long  SMAX  = (long)BB * HH * TCP1 * TT;               // 557056 >= H*T*T
  float* sf   = (float*)alloc(SMAX * 4);                 // attention scores
  u16*  pbuf  = (u16*)alloc(SMAX * 2);                   // softmax probs (bf16)
  u16*  kc[2], *vc[2];                                   // hoisted cross K/V for COT loop
  kc[0] = (u16*)alloc(RMAX * EE * 2); vc[0] = (u16*)alloc(RMAX * EE * 2);
  kc[1] = (u16*)alloc(RMAX * EE * 2); vc[1] = (u16*)alloc(RMAX * EE * 2);
  float* ce   = (float*)alloc((long)BB * TCP1 * EE * 4);
  u16*  ceEb  = (u16*)alloc((long)BB * TCC * EE * 2);
  float* lnf32 = (float*)alloc(RMAX * EE * 4);
  u16*  meanb = (u16*)alloc((long)BB * EE * 2);
  int*  ids   = (int*)alloc((long)BB * TCP1 * 4);

  float* outF      = (float*)d_out;
  float* outTok    = outF;                                   // [B,TC]
  float* outCotLog = outF + BB * TCC;                        // [B,TC,CV]
  float* outEncLog = outF + BB * TCC + (long)BB * TCC * CVV; // [B,V]

  // ---- GEMM wrappers ----
  auto gemmF = [&](const u16* A, long sAm, long sAh, long sAb,
                   const u16* Bm, long sBn, long sBh, long sBb,
                   float* C, long sCm, long sCn, long sCh, long sCb,
                   const float* Res, u16* Cb,
                   int M, int N, int K, int nH, int nB, int act) {
    if ((M % 32 == 0) && (N % 32 == 0)) {
      dim3 grid(N / 32, M / 32, nH * nB);
      k_gemm_f2<<<grid, 32, 0, stream>>>(A, sAm, sAh, sAb, Bm, sBn, sBh, sBb,
                                         C, sCm, sCn, sCh, sCb, Res, Cb, M, N, K, nH, act);
    } else {
      dim3 grid(N / 16, M / 16, nH * nB);
      k_gemm_f<<<grid, 32, 0, stream>>>(A, sAm, sAh, sAb, Bm, sBn, sBh, sBb,
                                        C, sCm, sCn, sCh, sCb, Res, Cb, M, N, K, nH, act);
    }
  };
  auto gemmG = [&](const u16* A, long sAm, long sAk, long sAh, long sAb,
                   const u16* Bm, long sBk, long sBn, long sBh, long sBb,
                   float* C, long sCm, long sCn, long sCh, long sCb,
                   const float* Res, u16* Cb,
                   int M, int N, int K, int nH, int nB, int act) {
    dim3 grid((N + 15) / 16, (M + 15) / 16, nH * nB);
    k_gemm<<<grid, 32, 0, stream>>>(A, sAm, sAk, sAh, sAb, Bm, sBk, sBn, sBh, sBb,
                                    C, sCm, sCn, sCh, sCb, Res, Cb, M, N, K, nH, act);
  };
  // X[M,K] @ W (WT[N][K]): M%16==0, K%32==0, N%16==0 always -> fast
  auto gemmW = [&](const u16* A, long sAm, const u16* WT, int N, int K, int M,
                   float* C, const float* Res, u16* Cb, int act) {
    gemmF(A, sAm, 0, 0, WT, K, 0, 0, C, N, 1, 0, 0, Res, Cb, M, N, K, 1, 1, act);
  };
  // V projection with transposed output Vt[b][h*D+d][t]; batched over b
  auto vproj = [&](const u16* src, const u16* WT, u16* Vt, int Tk) {
    if (Tk % 16 == 0)
      gemmF(src, EE, 0, (long)Tk * EE, WT, EE, 0, 0,
            nullptr, 1, Tk, 0, (long)HH * DD * Tk, nullptr, Vt, Tk, EE, EE, 1, BB, 0);
    else
      gemmG(src, EE, 1, 0, (long)Tk * EE, WT, 1, EE, 0, 0,
            nullptr, 1, Tk, 0, (long)HH * DD * Tk, nullptr, Vt, Tk, EE, EE, 1, BB, 0);
  };
  auto ln = [&](const float* x, const float* g, float* oF, u16* oB, int rows) {
    k_ln<<<dim3(rows), dim3(256), 0, stream>>>(x, g, oF, oB);
  };

  const float scl = 0.17677669529663687f;  // 1/sqrt(D)
  // Q[B,Tq,H,D], K[B,Tk,H,D], Vt[B,H*D,Tk] -> O[B,Tq,H,D]. big: per-b loop reusing sf.
  auto attn_core = [&](const u16* Q, const u16* Kx, const u16* Vt, u16* O,
                       int Tq, int Tk, int causal, int big) {
    if (big) {   // Tq = Tk = 256, all fast (32x32 tiles)
      for (int b = 0; b < BB; b++) {
        const u16* Qb = Q  + (long)b * Tq * HH * DD;
        const u16* Kb = Kx + (long)b * Tk * HH * DD;
        const u16* Vb = Vt + (long)b * HH * DD * Tk;
        u16* Ob = O + (long)b * Tq * HH * DD;
        gemmF(Qb, HH * DD, DD, 0, Kb, HH * DD, DD, 0,
              sf, Tk, 1, (long)Tq * Tk, 0, nullptr, nullptr, Tq, Tk, DD, HH, 1, 0);
        k_softmax<<<dim3(HH * Tq), dim3(256), 0, stream>>>(sf, pbuf, Tk, Tq, scl, causal);
        gemmF(pbuf, Tk, (long)Tq * Tk, 0, Vb, Tk, (long)DD * Tk, 0,
              nullptr, HH * DD, 1, DD, 0, nullptr, Ob, Tq, DD, Tk, HH, 1, 0);
      }
    } else {
      bool fS = (Tq % 16 == 0) && (Tk % 16 == 0);
      bool fA = (Tq % 16 == 0) && (Tk % 32 == 0);
      if (fS)
        gemmF(Q, HH * DD, DD, (long)Tq * HH * DD, Kx, HH * DD, DD, (long)Tk * HH * DD,
              sf, Tk, 1, (long)Tq * Tk, (long)HH * Tq * Tk, nullptr, nullptr,
              Tq, Tk, DD, HH, BB, 0);
      else
        gemmG(Q, HH * DD, 1, DD, (long)Tq * HH * DD, Kx, 1, HH * DD, DD, (long)Tk * HH * DD,
              sf, Tk, 1, (long)Tq * Tk, (long)HH * Tq * Tk, nullptr, nullptr,
              Tq, Tk, DD, HH, BB, 0);
      k_softmax<<<dim3(BB * HH * Tq), dim3(256), 0, stream>>>(sf, pbuf, Tk, Tq, scl, causal);
      if (fA)
        gemmF(pbuf, Tk, (long)Tq * Tk, (long)HH * Tq * Tk, Vt, Tk, (long)DD * Tk, (long)HH * DD * Tk,
              nullptr, HH * DD, 1, DD, (long)Tq * HH * DD, nullptr, O, Tq, DD, Tk, HH, BB, 0);
      else
        gemmG(pbuf, Tk, 1, (long)Tq * Tk, (long)HH * Tq * Tk, Vt, 1, Tk, (long)DD * Tk, (long)HH * DD * Tk,
              nullptr, HH * DD, 1, DD, (long)Tq * HH * DD, nullptr, O, Tq, DD, Tk, HH, BB, 0);
    }
  };

  auto tlayer = [&](float* x, const TLayerP& p, const TLayerB& w, int Tq, int big, u16* finalB) {
    int R = BB * Tq;
    ln(x, p.ln1, nullptr, hb, R);
    gemmW(hb, EE, w.a.q, EE, EE, R, nullptr, nullptr, qb, 0);
    gemmW(hb, EE, w.a.k, EE, EE, R, nullptr, nullptr, kbuf, 0);
    vproj(hb, w.a.v, vbuf, Tq);
    attn_core(qb, kbuf, vbuf, obuf, Tq, Tq, 0, big);
    gemmW(obuf, EE, w.a.o, EE, EE, R, x, x, nullptr, 0);       // x += attn
    ln(x, p.ln2, nullptr, hb, R);
    gemmW(hb, EE, w.w1, FF, EE, R, nullptr, nullptr, hid, 1);  // GELU
    gemmW(hid, FF, w.w2, EE, FF, R, x, x, finalB, 0);          // x += mlp
  };

  auto clayer_cot = [&](float* x, const CLayerP& p, const CLayerB& w, u16* kcr, u16* vcr) {
    int Tq = TCP1, R = BB * Tq;
    ln(x, p.ln1, nullptr, hb, R);
    gemmW(hb, EE, w.sa.q, EE, EE, R, nullptr, nullptr, qb, 0);
    gemmW(hb, EE, w.sa.k, EE, EE, R, nullptr, nullptr, kbuf, 0);
    vproj(hb, w.sa.v, vbuf, Tq);
    attn_core(qb, kbuf, vbuf, obuf, Tq, Tq, 1, 0);             // causal self
    gemmW(obuf, EE, w.sa.o, EE, EE, R, x, x, nullptr, 0);
    ln(x, p.ln2, nullptr, hb, R);
    gemmW(hb, EE, w.ca.q, EE, EE, R, nullptr, nullptr, qb, 0);
    attn_core(qb, kcr, vcr, obuf, Tq, TT, 0, 0);               // cross (hoisted K/V)
    gemmW(obuf, EE, w.ca.o, EE, EE, R, x, x, nullptr, 0);
    ln(x, p.ln3, nullptr, hb, R);
    gemmW(hb, EE, w.w1, FF, EE, R, nullptr, nullptr, hid, 1);
    gemmW(hid, FF, w.w2, EE, FF, R, x, x, nullptr, 0);
  };

  auto clayer_enc = [&](float* x, const CLayerP& p, const CLayerB& w) {
    int R = BB * TT;
    ln(x, p.ln1, nullptr, hb, R);
    gemmW(hb, EE, w.sa.q, EE, EE, R, nullptr, nullptr, qb, 0);
    gemmW(hb, EE, w.sa.k, EE, EE, R, nullptr, nullptr, kbuf, 0);
    vproj(hb, w.sa.v, vbuf, TT);
    attn_core(qb, kbuf, vbuf, obuf, TT, TT, 0, 1);             // big self-attn
    gemmW(obuf, EE, w.sa.o, EE, EE, R, x, x, nullptr, 0);
    ln(x, p.ln2, nullptr, hb, R);
    gemmW(hb, EE, w.ca.q, EE, EE, R, nullptr, nullptr, qb, 0);
    gemmW(ceEb, EE, w.ca.k, EE, EE, BB * TCC, nullptr, nullptr, kbuf, 0);
    vproj(ceEb, w.ca.v, vbuf, TCC);
    attn_core(qb, kbuf, vbuf, obuf, TT, TCC, 0, 0);            // cross to cot enc
    gemmW(obuf, EE, w.ca.o, EE, EE, R, x, x, nullptr, 0);
    ln(x, p.ln3, nullptr, hb, R);
    gemmW(hb, EE, w.w1, FF, EE, R, nullptr, nullptr, hid, 1);
    gemmW(hid, FF, w.w2, EE, FF, R, x, x, nullptr, 0);
  };

  // ================= COT module =================
  k_embed<<<dim3(BB * TT), dim3(256), 0, stream>>>(inputs, TT, 0, cot_tok, cot_pos, TT, xE, nullptr);
  tlayer(xE, cin[0], cinB[0], TT, 1, nullptr);
  tlayer(xE, cin[1], cinB[1], TT, 1, xEb);   // also emit bf16 of final x

  // hoist cross K/V (depend only on x) out of the 16-step loop
  for (int l = 0; l < 2; l++) {
    gemmW(xEb, EE, ccrB[l].ca.k, EE, EE, BB * TT, nullptr, nullptr, kc[l], 0);
    vproj(xEb, ccrB[l].ca.v, vc[l], TT);
  }

  k_init_ids<<<dim3(2), dim3(256), 0, stream>>>(ids);

  for (int i = 0; i < TCC; i++) {
    k_embed<<<dim3(BB * TCP1), dim3(256), 0, stream>>>(ids, TCP1, 0, cot_ctok, cot_cpos, TCP1, ce, nullptr);
    for (int l = 0; l < 2; l++) clayer_cot(ce, ccr[l], ccrB[l], kc[l], vc[l]);
    ln(ce, cot_lnf, nullptr, hb, BB * TCP1);
    // logits for decode position i only -> outCotLog[:, i, :]
    gemmF(hb + (long)i * EE, (long)TCP1 * EE, 0, 0,
          cotHeadT, EE, 0, 0,
          outCotLog + (long)i * CVV, (long)TCC * CVV, 1, 0, 0,
          nullptr, nullptr, BB, CVV, EE, 1, 1, 0);
    k_argmax<<<dim3(BB), dim3(256), 0, stream>>>(outCotLog + (long)i * CVV, (long)TCC * CVV,
                                                 ids, TCP1, i + 1, outTok + i, TCC);
  }

  // ================= Encoder =================
  k_embed<<<dim3(BB * TCC), dim3(256), 0, stream>>>(ids, TCP1, 1, enc_ctok, enc_cpos, TCC, ce, nullptr);
  tlayer(ce, ecot[0], ecotB[0], TCC, 0, nullptr);
  tlayer(ce, ecot[1], ecotB[1], TCC, 0, ceEb);   // bf16 of cot encoding

  k_embed<<<dim3(BB * TT), dim3(256), 0, stream>>>(inputs, TT, 0, enc_tok, enc_pos, TT, xE, nullptr);
  clayer_enc(xE, ecr[0], ecrB[0]);
  clayer_enc(xE, ecr[1], ecrB[1]);

  ln(xE, enc_lnf, lnf32, nullptr, BB * TT);
  k_mean<<<dim3(BB), dim3(256), 0, stream>>>(lnf32, meanb, TT);
  gemmW(meanb, EE, encHeadT, VV, EE, BB, outEncLog, nullptr, nullptr, 0);
}
